// MinimalQuantumCell_18279380812188
// MI455X (gfx1250) — compile-verified, matrix operand order checked by reading, not run on hardware
//
#include <hip/hip_runtime.h>
#include <math.h>

// MinimalQuantumCell on MI455X (gfx1250, wave32).
// GEMM M=4096,K=256,N=8192 (N = hid*16+state, contiguous) as bf16 WMMA / f32 acc,
// fused epilogue: tanh(ic + 0.9*h) via v_tanh_f32 (CDNA5 TRANS op), 128MB state
// store (non-temporal), per-16-state mean via 16-lane shfl_xor butterfly.
// Roofline: ~276 MB HBM -> ~12 us @ 23.3 TB/s; GEMM needs only ~1.5 TF effective.
// x(4MB)+W(8MB) stay L2-hot (192MB L2); h/state streams are NT-hinted.
// Wave tile = 32x64 (2 M-tiles x 4 N-tiles): 8 WMMAs per K-chunk off 6 fragment
// conversions, keeping VALU comfortably below the memory floor.

typedef __attribute__((ext_vector_type(16))) __bf16 v16bf;
typedef __attribute__((ext_vector_type(8)))  float  v8f;

#define BATCH  4096
#define IN_DIM 256
#define HID    512
#define STATES 16
#define NDIM   (HID * STATES)   // 8192

// CDNA5 transcendental tanh. Trailing v_nop satisfies the TRANS hazard rule
// (one independent op or NOP between a trans op and any use of its result) even if the
// compiler schedules the consumer immediately after the asm block.
__device__ __forceinline__ float fast_tanh(float v) {
    float r;
    asm volatile("v_tanh_f32 %0, %1\n\tv_nop" : "=v"(r) : "v"(v));
    return r;
}

__global__ __launch_bounds__(256)
void qcell_wmma_kernel(const float* __restrict__ x,
                       const float* __restrict__ hq,
                       const float* __restrict__ W,
                       float* __restrict__ out_mean,   // [BATCH, HID]
                       float* __restrict__ out_state)  // [BATCH, HID, STATES]
{
    const int lane = threadIdx.x & 31;
    const int wave = threadIdx.x >> 5;
    const int l15  = lane & 15;
    const int half = lane >> 4;            // 0: lanes 0-15, 1: lanes 16-31

    const int wy = wave >> 2;              // 0..1  (M direction)
    const int wx = wave & 3;               // 0..3  (N direction)

    const int m0 = blockIdx.x * 64 + wy * 32;    // wave: rows m0..m0+31 (2 tiles)
    const int n0 = blockIdx.y * 256 + wx * 64;   // wave: cols n0..n0+63 (4 tiles)

    v8f acc[2][4] = {};

    for (int k0 = 0; k0 < IN_DIM; k0 += 32) {
        const int kb = k0 + half * 8;            // lane-half K base

        // ---- A fragments: 2 M-tiles, x[row, kb..kb+7] and x[row, kb+16..kb+23]
        v16bf af[2];
        #pragma unroll
        for (int mt = 0; mt < 2; ++mt) {
            const int arow = m0 + mt * 16 + l15;
            const float4* ap = (const float4*)(x + (size_t)arow * IN_DIM + kb);
            float4 a0 = ap[0];
            float4 a1 = ap[1];
            const float4* ap2 = (const float4*)(x + (size_t)arow * IN_DIM + kb + 16);
            float4 a2 = ap2[0];
            float4 a3 = ap2[1];
            af[mt][0]=(__bf16)a0.x;  af[mt][1]=(__bf16)a0.y;
            af[mt][2]=(__bf16)a0.z;  af[mt][3]=(__bf16)a0.w;
            af[mt][4]=(__bf16)a1.x;  af[mt][5]=(__bf16)a1.y;
            af[mt][6]=(__bf16)a1.z;  af[mt][7]=(__bf16)a1.w;
            af[mt][8]=(__bf16)a2.x;  af[mt][9]=(__bf16)a2.y;
            af[mt][10]=(__bf16)a2.z; af[mt][11]=(__bf16)a2.w;
            af[mt][12]=(__bf16)a3.x; af[mt][13]=(__bf16)a3.y;
            af[mt][14]=(__bf16)a3.z; af[mt][15]=(__bf16)a3.w;
        }

        // ---- 4 B fragments; each feeds both M-tiles' WMMAs ----
        #pragma unroll
        for (int j = 0; j < 4; ++j) {
            const int bcol = n0 + j * 16 + l15;
            const float* bp = W + (size_t)kb * NDIM + bcol;
            v16bf bf;
            #pragma unroll
            for (int t = 0; t < 8; ++t) {
                bf[t]     = (__bf16)bp[(size_t)t * NDIM];          // K = kb+t
                bf[t + 8] = (__bf16)bp[(size_t)(t + 16) * NDIM];   // K = kb+16+t
            }
            acc[0][j] = __builtin_amdgcn_wmma_f32_16x16x32_bf16(
                            false, af[0], false, bf, (short)0, acc[0][j], false, false);
            acc[1][j] = __builtin_amdgcn_wmma_f32_16x16x32_bf16(
                            false, af[1], false, bf, (short)0, acc[1][j], false, false);
        }
    }

    // ---- Epilogue: tanh(acc + 0.9*h), stream state out, mean over 16 states ----
    #pragma unroll
    for (int mt = 0; mt < 2; ++mt) {
        const int rbase = m0 + mt * 16 + half * 8;   // D layout: half1 holds M=v+8
        #pragma unroll
        for (int j = 0; j < 4; ++j) {
            const int ncol = n0 + j * 16 + l15;
            const int hidx = (n0 + j * 16) >> 4;     // one h per 16-wide tile
            #pragma unroll
            for (int v = 0; v < 8; ++v) {
                const int row = rbase + v;
                const size_t off = (size_t)row * NDIM + ncol;
                float h = __builtin_nontemporal_load(hq + off);     // 128MB stream
                float t = fast_tanh(acc[mt][j][v] + 0.9f * h);
                __builtin_nontemporal_store(t, out_state + off);    // 128MB stream
                // mean over states = reduce this row's 16 lanes (per half)
                float s = t;
                s += __shfl_xor(s, 1, 16);
                s += __shfl_xor(s, 2, 16);
                s += __shfl_xor(s, 4, 16);
                s += __shfl_xor(s, 8, 16);
                if (l15 == 0)
                    out_mean[(size_t)row * HID + hidx] = s * (1.0f / 16.0f);
            }
        }
    }
}

extern "C" void kernel_launch(void* const* d_in, const int* in_sizes, int n_in,
                              void* d_out, int out_size, void* d_ws, size_t ws_size,
                              hipStream_t stream) {
    const float* x  = (const float*)d_in[0];   // [4096, 256]
    const float* hq = (const float*)d_in[1];   // [4096, 512, 16]
    const float* W  = (const float*)d_in[2];   // [256, 512, 16]

    float* out_mean  = (float*)d_out;                          // [4096, 512]
    float* out_state = out_mean + (size_t)BATCH * HID;         // [4096, 512, 16]

    dim3 grid(BATCH / 64, NDIM / 256);   // (64, 32)
    dim3 block(256);                     // 8 waves; each owns a 32x64 C tile
    qcell_wmma_kernel<<<grid, block, 0, stream>>>(x, hq, W, out_mean, out_state);
}
